// LSTM_ATT_Model_44049184587870
// MI455X (gfx1250) — compile-verified, matrix operand order checked by reading, not run on hardware
//
#include <hip/hip_runtime.h>

typedef __bf16 bf16;
typedef __attribute__((ext_vector_type(16))) __bf16 v16bf;
typedef __attribute__((ext_vector_type(8)))  __bf16 v8bf;
typedef __attribute__((ext_vector_type(8)))  float  v8f;
typedef __attribute__((ext_vector_type(4)))  int    v4i;

#define GLOBAL_AS __attribute__((address_space(1)))
#define LDS_AS    __attribute__((address_space(3)))

constexpr int T    = 400;
constexpr int B    = 256;
constexpr int F    = 40;
constexpr int HID  = 768;
constexpr int PROJ = 256;
constexpr int ATT  = 128;
constexpr int G4   = 4 * HID;   // 3072
constexpr int KW   = 512;       // padded concat K (40+256 -> 512, 256+256 = 512)
constexpr int LPADW = KW + 8;   // LDS row stride (bf16) for gates A-slab (bank-spread)
constexpr int LPADH = HID + 8;  // LDS row stride (bf16) for proj A-slab

#if defined(__has_builtin)
#if __has_builtin(__builtin_amdgcn_global_load_async_to_lds_b128) && \
    __has_builtin(__builtin_amdgcn_s_wait_asynccnt)
#define HAVE_ASYNC_LDS 1
#endif
#endif
#ifndef HAVE_ASYNC_LDS
#define HAVE_ASYNC_LDS 0
#endif

__device__ __forceinline__ float sigmoidf_(float x) { return 1.0f / (1.0f + __expf(-x)); }

// Stage `rows` x `cols` bf16 slab (row-major, global stride = cols) into LDS with
// row stride `lstride`, using CDNA5 async global->LDS when available.
__device__ __forceinline__ void stage_slab(const bf16* __restrict__ g, bf16* l,
                                           int rows, int cols, int lstride) {
  const int cpr = cols / 8;  // 16-byte chunks per row
  const int total = rows * cpr;
#if HAVE_ASYNC_LDS
  for (int c = threadIdx.x; c < total; c += blockDim.x) {
    int r = c / cpr, cc = c % cpr;
    __builtin_amdgcn_global_load_async_to_lds_b128(
        (GLOBAL_AS v4i*)(g + (long long)r * cols + cc * 8),
        (LDS_AS v4i*)(l + r * lstride + cc * 8), 0, 0);
  }
  __builtin_amdgcn_s_wait_asynccnt(0);
  __syncthreads();
#else
  for (int c = threadIdx.x; c < total; c += blockDim.x) {
    int r = c / cpr, cc = c % cpr;
    *(v8bf*)(l + r * lstride + cc * 8) = *(const v8bf*)(g + (long long)r * cols + cc * 8);
  }
  __syncthreads();
#endif
}

// ---------------------------------------------------------------------------
// Pack fp32 [Krows x N] row-major matrix into bf16 WMMA B-operand tiles.
// Packed layout: ((kt*NT + nt)*32 + lane)*16 + i, where for lane:
//   half = lane>>4, n = nt*16 + (lane&15)
//   i in [0,8):  k = kt*32 + half*8 + i
//   i in [8,16): k = kt*32 + 16 + half*8 + (i-8)
// Rows k >= Krows are zero-padded (handles layer-0 K=296 -> 512).
// ---------------------------------------------------------------------------
__global__ void pack_b_kernel(const float* __restrict__ src, bf16* __restrict__ dst,
                              int Krows, int N, int KT) {
  const int NT = N >> 4;
  const long long total = (long long)KT * NT * 32 * 16;
  long long idx = (long long)blockIdx.x * blockDim.x + threadIdx.x;
  if (idx >= total) return;
  int i    = (int)(idx & 15);
  int lane = (int)((idx >> 4) & 31);
  long long tile = idx >> 9;
  int nt = (int)(tile % NT);
  int kt = (int)(tile / NT);
  int kh = lane >> 4;
  int n  = nt * 16 + (lane & 15);
  int k  = kt * 32 + kh * 8 + ((i < 8) ? i : (i - 8) + 16);
  float v = (k < Krows) ? src[(long long)k * N + n] : 0.0f;
  dst[idx] = (bf16)v;
}

// Fill concat buffer Acat [T][B][KW] bf16: cols [0,Din) = layer input, rest 0.
__global__ void prep_acat_kernel(const float* __restrict__ in, bf16* __restrict__ acat, int Din) {
  long long total = (long long)T * B * KW;
  long long idx = (long long)blockIdx.x * blockDim.x + threadIdx.x;
  if (idx >= total) return;
  int col = (int)(idx % KW);
  long long tb = idx / KW;
  float v = (col < Din) ? in[tb * Din + col] : 0.0f;
  acat[idx] = (bf16)v;
}

__global__ void zero_f32_kernel(float* __restrict__ p, long long n) {
  long long idx = (long long)blockIdx.x * blockDim.x + threadIdx.x;
  if (idx < n) p[idx] = 0.0f;
}

// ---------------------------------------------------------------------------
// Per-step gate GEMM + LSTM cell update.
//   z = [xt|m] (256x512) @ W (512x3072) + b ; gates i,j,f,o (col blocks of 768)
//   c = sig(f+1)*c + sig(i)*tanh(j) ; h = sig(o)*tanh(c)
// Block (8 waves) covers 32 rows x 128 h-cols; A slab (32x512) staged in LDS.
// Each wave: 32x16 tile = 2 row-tiles x 4 gates = 8 accs; 8 wmma per k-chunk,
// 16 k-chunks => 128 v_wmma per wave.
// grid = 48 blocks x 256 threads.
// ---------------------------------------------------------------------------
__global__ __launch_bounds__(256) void lstm_gates_kernel(
    const bf16* __restrict__ Acat_t,  // [B, KW] bf16 (this timestep)
    const bf16* __restrict__ Wpk,     // packed [16][192][32][16]
    const float* __restrict__ bias,   // [3072]
    float* __restrict__ C,            // [B, HID] fp32 state (in/out)
    bf16* __restrict__ H)             // [B, HID] bf16 out
{
  __shared__ bf16 As[32 * LPADW];  // 33.3 KB

  const int NT   = G4 >> 4;  // 192
  const int lane = threadIdx.x & 31;
  const int wave = threadIdx.x >> 5;               // 0..7
  const int rowbase  = (blockIdx.x / 6) * 32;      // 8 row groups
  const int colgroup = blockIdx.x % 6;             // 6 col groups of 128
  const int tc   = colgroup * 8 + wave;            // 0..47 col tile
  const int kh   = lane >> 4;
  const int lr   = lane & 15;

  stage_slab(Acat_t + (long long)rowbase * KW, As, 32, KW, LPADW);

  const bf16* arow0 = As + lr * LPADW;         // rows 0..15 of slab
  const bf16* arow1 = As + (16 + lr) * LPADW;  // rows 16..31 of slab

  v8f acc[2][4];
#pragma unroll
  for (int rt = 0; rt < 2; ++rt)
#pragma unroll
    for (int g = 0; g < 4; ++g) acc[rt][g] = (v8f){};

#pragma unroll
  for (int kt = 0; kt < KW / 32; ++kt) {
    v8bf a00 = *(const v8bf*)(arow0 + kt * 32 + kh * 8);
    v8bf a01 = *(const v8bf*)(arow0 + kt * 32 + 16 + kh * 8);
    v8bf a10 = *(const v8bf*)(arow1 + kt * 32 + kh * 8);
    v8bf a11 = *(const v8bf*)(arow1 + kt * 32 + 16 + kh * 8);
    v16bf a0 = __builtin_shufflevector(a00, a01, 0, 1, 2, 3, 4, 5, 6, 7,
                                       8, 9, 10, 11, 12, 13, 14, 15);
    v16bf a1 = __builtin_shufflevector(a10, a11, 0, 1, 2, 3, 4, 5, 6, 7,
                                       8, 9, 10, 11, 12, 13, 14, 15);
    const bf16* wbase = Wpk + (((long long)kt * NT) * 32 + lane) * 16;
    v16bf b0 = *(const v16bf*)(wbase + (long long)(0 * 48 + tc) * 512);
    v16bf b1 = *(const v16bf*)(wbase + (long long)(1 * 48 + tc) * 512);
    v16bf b2 = *(const v16bf*)(wbase + (long long)(2 * 48 + tc) * 512);
    v16bf b3 = *(const v16bf*)(wbase + (long long)(3 * 48 + tc) * 512);
    acc[0][0] = __builtin_amdgcn_wmma_f32_16x16x32_bf16(false, a0, false, b0, (short)0, acc[0][0], false, false);
    acc[1][0] = __builtin_amdgcn_wmma_f32_16x16x32_bf16(false, a1, false, b0, (short)0, acc[1][0], false, false);
    acc[0][1] = __builtin_amdgcn_wmma_f32_16x16x32_bf16(false, a0, false, b1, (short)0, acc[0][1], false, false);
    acc[1][1] = __builtin_amdgcn_wmma_f32_16x16x32_bf16(false, a1, false, b1, (short)0, acc[1][1], false, false);
    acc[0][2] = __builtin_amdgcn_wmma_f32_16x16x32_bf16(false, a0, false, b2, (short)0, acc[0][2], false, false);
    acc[1][2] = __builtin_amdgcn_wmma_f32_16x16x32_bf16(false, a1, false, b2, (short)0, acc[1][2], false, false);
    acc[0][3] = __builtin_amdgcn_wmma_f32_16x16x32_bf16(false, a0, false, b3, (short)0, acc[0][3], false, false);
    acc[1][3] = __builtin_amdgcn_wmma_f32_16x16x32_bf16(false, a1, false, b3, (short)0, acc[1][3], false, false);
  }

  // D layout: lane holds col n = lr, rows = base + r + 8*(lane>>4), r = 0..7
  const int n = tc * 16 + lr;
  const float bi  = bias[0 * HID + n];
  const float bj  = bias[1 * HID + n];
  const float bfv = bias[2 * HID + n];
  const float bo  = bias[3 * HID + n];
#pragma unroll
  for (int rt = 0; rt < 2; ++rt) {
#pragma unroll
    for (int r = 0; r < 8; ++r) {
      int row = rowbase + rt * 16 + r + kh * 8;
      long long ci = (long long)row * HID + n;
      float cold = C[ci];
      float ig = sigmoidf_(acc[rt][0][r] + bi);
      float jg = tanhf(acc[rt][1][r] + bj);
      float fg = sigmoidf_(acc[rt][2][r] + bfv + 1.0f);
      float og = sigmoidf_(acc[rt][3][r] + bo);
      float cnew = fg * cold + ig * jg;
      C[ci] = cnew;
      H[ci] = (bf16)(og * tanhf(cnew));
    }
  }
}

// ---------------------------------------------------------------------------
// Per-step projection: m = h (256x768) @ P (768x256).
// Block (8 waves) covers 32 rows x 128 cols; h slab (32x768) staged in LDS.
// Each wave: 32x16 tile (2 accs); 24 k-chunks => 48 wmma per wave.
// grid = 16 blocks x 256 threads.
// ---------------------------------------------------------------------------
__global__ __launch_bounds__(256) void lstm_proj_kernel(
    const bf16* __restrict__ H,    // [B, HID] bf16
    const bf16* __restrict__ Ppk,  // packed [24][16][32][16]
    float* __restrict__ Mout_t,    // [B, PROJ] fp32 (this timestep slice)
    bf16* __restrict__ AcatNext,   // [B, KW] bf16 (t+1) or nullptr
    int mslot)                     // = Din (concat offset of m)
{
  __shared__ bf16 Hs[32 * LPADH];  // 48.5 KB

  const int NT   = PROJ >> 4;  // 16
  const int lane = threadIdx.x & 31;
  const int wave = threadIdx.x >> 5;               // 0..7
  const int rowbase  = (blockIdx.x / 2) * 32;      // 8 row groups
  const int colgroup = blockIdx.x % 2;             // 2 col groups of 128
  const int tc   = colgroup * 8 + wave;            // 0..15 col tile
  const int kh   = lane >> 4;
  const int lr   = lane & 15;

  stage_slab(H + (long long)rowbase * HID, Hs, 32, HID, LPADH);

  const bf16* arow0 = Hs + lr * LPADH;
  const bf16* arow1 = Hs + (16 + lr) * LPADH;

  v8f acc0 = {}, acc1 = {};
#pragma unroll
  for (int kt = 0; kt < HID / 32; ++kt) {  // 24
    v8bf a00 = *(const v8bf*)(arow0 + kt * 32 + kh * 8);
    v8bf a01 = *(const v8bf*)(arow0 + kt * 32 + 16 + kh * 8);
    v8bf a10 = *(const v8bf*)(arow1 + kt * 32 + kh * 8);
    v8bf a11 = *(const v8bf*)(arow1 + kt * 32 + 16 + kh * 8);
    v16bf a0 = __builtin_shufflevector(a00, a01, 0, 1, 2, 3, 4, 5, 6, 7,
                                       8, 9, 10, 11, 12, 13, 14, 15);
    v16bf a1 = __builtin_shufflevector(a10, a11, 0, 1, 2, 3, 4, 5, 6, 7,
                                       8, 9, 10, 11, 12, 13, 14, 15);
    v16bf b = *(const v16bf*)(Ppk + (((long long)kt * NT + tc) * 32 + lane) * 16);
    acc0 = __builtin_amdgcn_wmma_f32_16x16x32_bf16(false, a0, false, b, (short)0, acc0, false, false);
    acc1 = __builtin_amdgcn_wmma_f32_16x16x32_bf16(false, a1, false, b, (short)0, acc1, false, false);
  }

  const int n = tc * 16 + lr;
#pragma unroll
  for (int rt = 0; rt < 2; ++rt) {
    const v8f& acc = rt ? acc1 : acc0;
#pragma unroll
    for (int r = 0; r < 8; ++r) {
      int row = rowbase + rt * 16 + r + kh * 8;
      float m = acc[r];
      Mout_t[(long long)row * PROJ + n] = m;
      if (AcatNext) AcatNext[(long long)row * KW + mslot + n] = (bf16)m;
    }
  }
}

// ---------------------------------------------------------------------------
// Attention: vu[t,b] = sum_a tanh(M2[t,b,:] . w_att[:,a] + b_att[a]) * u_att[a]
// ---------------------------------------------------------------------------
__global__ __launch_bounds__(128) void att_vu_kernel(
    const float* __restrict__ M2, const float* __restrict__ w_att,
    const float* __restrict__ b_att, const float* __restrict__ u_att,
    float* __restrict__ vu) {
  __shared__ float sm[PROJ];
  __shared__ float red[128];
  long long tb = blockIdx.x;
  int a = threadIdx.x;
  sm[a]       = M2[tb * PROJ + a];
  sm[a + 128] = M2[tb * PROJ + a + 128];
  __syncthreads();
  float dot = 0.0f;
#pragma unroll 8
  for (int p = 0; p < PROJ; ++p) dot += sm[p] * w_att[p * ATT + a];
  red[a] = tanhf(dot + b_att[a]) * u_att[a];
  __syncthreads();
  for (int s = 64; s > 0; s >>= 1) {
    if (a < s) red[a] += red[a + s];
    __syncthreads();
  }
  if (a == 0) vu[tb] = red[0];
}

// softmax over t per batch column + weighted pooling. one block per b.
__global__ __launch_bounds__(256) void att_pool_kernel(
    const float* __restrict__ M2, const float* __restrict__ vu,
    float* __restrict__ out_att) {
  int b = blockIdx.x, tid = threadIdx.x;
  __shared__ float red[256];
  float mx = -1e30f;
  for (int t = tid; t < T; t += 256) mx = fmaxf(mx, vu[(long long)t * B + b]);
  red[tid] = mx;
  __syncthreads();
  for (int s = 128; s > 0; s >>= 1) {
    if (tid < s) red[tid] = fmaxf(red[tid], red[tid + s]);
    __syncthreads();
  }
  mx = red[0];
  __syncthreads();
  float se = 0.0f;
  for (int t = tid; t < T; t += 256) se += __expf(vu[(long long)t * B + b] - mx);
  red[tid] = se;
  __syncthreads();
  for (int s = 128; s > 0; s >>= 1) {
    if (tid < s) red[tid] += red[tid + s];
    __syncthreads();
  }
  float inv = 1.0f / red[0];
  float acc = 0.0f;  // tid == p
  for (int t = 0; t < T; ++t) {
    float al = __expf(vu[(long long)t * B + b] - mx) * inv;
    acc += al * M2[((long long)t * B + b) * PROJ + tid];
  }
  out_att[b * PROJ + tid] = acc;
}

// embedded = l2-normalized last timestep. one block per b.
__global__ __launch_bounds__(256) void att_last_kernel(
    const float* __restrict__ M2, float* __restrict__ out_emb) {
  int b = blockIdx.x, p = threadIdx.x;
  __shared__ float red[256];
  float v = M2[(((long long)(T - 1)) * B + b) * PROJ + p];
  red[p] = v * v;
  __syncthreads();
  for (int s = 128; s > 0; s >>= 1) {
    if (p < s) red[p] += red[p + s];
    __syncthreads();
  }
  float rs = rsqrtf(fmaxf(red[0], 1e-12f));
  out_emb[b * PROJ + p] = v * rs;
}

// ---------------------------------------------------------------------------
extern "C" void kernel_launch(void* const* d_in, const int* in_sizes, int n_in,
                              void* d_out, int out_size, void* d_ws, size_t ws_size,
                              hipStream_t stream) {
  (void)in_sizes; (void)n_in; (void)out_size; (void)ws_size;
  const float* x     = (const float*)d_in[0];
  const float* W[3]  = {(const float*)d_in[1], (const float*)d_in[4], (const float*)d_in[7]};
  const float* bb[3] = {(const float*)d_in[2], (const float*)d_in[5], (const float*)d_in[8]};
  const float* P[3]  = {(const float*)d_in[3], (const float*)d_in[6], (const float*)d_in[9]};
  const float* w_att = (const float*)d_in[10];
  const float* b_att = (const float*)d_in[11];
  const float* u_att = (const float*)d_in[12];
  float* out = (float*)d_out;

  char* ws = (char*)d_ws;
  size_t off = 0;
  auto take = [&](size_t bytes) -> char* {
    char* p = ws + off;
    off = (off + bytes + 255) & ~(size_t)255;
    return p;
  };
  bf16*  Acat = (bf16*)take((size_t)T * B * KW * 2);    // concat A-matrix, all t
  float* Ma   = (float*)take((size_t)T * B * PROJ * 4); // layer output ping
  float* Mb   = (float*)take((size_t)T * B * PROJ * 4); // layer output pong
  bf16*  Wpk  = (bf16*)take((size_t)KW * G4 * 2);       // packed W (per layer)
  bf16*  Ppk  = (bf16*)take((size_t)HID * PROJ * 2);    // packed P (per layer)
  float* C    = (float*)take((size_t)B * HID * 4);      // cell state
  bf16*  Hb   = (bf16*)take((size_t)B * HID * 2);       // hidden (pre-projection)
  float* vu   = (float*)take((size_t)T * B * 4);        // attention logits

  const float* layer_in[3]  = {x, Ma, Mb};
  float*       layer_out[3] = {Ma, Mb, Ma};
  const int    din[3]       = {F, PROJ, PROJ};

  for (int l = 0; l < 3; ++l) {
    const int Din   = din[l];
    const int Krows = Din + PROJ;
    {
      long long tot = (long long)(KW / 32) * (G4 / 16) * 32 * 16;
      pack_b_kernel<<<(int)((tot + 255) / 256), 256, 0, stream>>>(W[l], Wpk, Krows, G4, KW / 32);
    }
    {
      long long tot = (long long)(HID / 32) * (PROJ / 16) * 32 * 16;
      pack_b_kernel<<<(int)((tot + 255) / 256), 256, 0, stream>>>(P[l], Ppk, HID, PROJ, HID / 32);
    }
    {
      long long tot = (long long)T * B * KW;
      prep_acat_kernel<<<(int)((tot + 255) / 256), 256, 0, stream>>>(layer_in[l], Acat, Din);
    }
    {
      long long n = (long long)B * HID;
      zero_f32_kernel<<<(int)((n + 255) / 256), 256, 0, stream>>>(C, n);
    }
    for (int t = 0; t < T; ++t) {
      lstm_gates_kernel<<<48, 256, 0, stream>>>(Acat + (long long)t * B * KW, Wpk, bb[l], C, Hb);
      bf16* acn = (t + 1 < T) ? (Acat + (long long)(t + 1) * B * KW) : nullptr;
      lstm_proj_kernel<<<16, 256, 0, stream>>>(Hb, Ppk, layer_out[l] + (long long)t * B * PROJ,
                                               acn, Din);
    }
  }

  const float* M2 = Ma;  // final layer outputs [T,B,PROJ]
  att_vu_kernel<<<T * B, 128, 0, stream>>>(M2, w_att, b_att, u_att, vu);
  att_pool_kernel<<<B, 256, 0, stream>>>(M2, vu, out + B * PROJ);  // att_embedded
  att_last_kernel<<<B, 256, 0, stream>>>(M2, out);                 // embedded
}
